// HydraAttention_19035295056603
// MI455X (gfx1250) — matrix operand order repaired and checked by reading, not compiled, not run on hardware
//
#include <hip/hip_runtime.h>

typedef __bf16 bf16;
typedef __attribute__((ext_vector_type(8)))  bf16     v8bf;
typedef __attribute__((ext_vector_type(16))) bf16     v16bf;
typedef __attribute__((ext_vector_type(8)))  float    v8f;
typedef __attribute__((ext_vector_type(4)))  unsigned u32x4;
typedef __attribute__((ext_vector_type(8)))  unsigned u32x8;

// Problem constants (from reference)
constexpr int B_  = 8;
constexpr int N_  = 16384;
constexpr int C_  = 256;
constexpr int NH_ = 8;
constexpr int CH_ = 32;
constexpr int K_  = 256;      // GEMM reduction dim (= C)
constexpr int HH_ = 128;
constexpr int WW_ = 128;
constexpr float SCALE_ = 0.1767766952966369f;   // 32^-0.5

// ---------------------------------------------------------------------------
// fp32 -> bf16 convert (grid-stride)
// ---------------------------------------------------------------------------
__global__ void k_f32_to_bf16(const float* __restrict__ in, bf16* __restrict__ out, int n) {
  int stride = gridDim.x * blockDim.x;
  for (int i = blockIdx.x * blockDim.x + threadIdx.x; i < n; i += stride)
    out[i] = (bf16)in[i];
}

__global__ void k_zero(float* __restrict__ p, int n) {
  int i = blockIdx.x * blockDim.x + threadIdx.x;
  if (i < n) p[i] = 0.0f;
}

// ---------------------------------------------------------------------------
// Pack a K_ x NC row-major fp32 weight into per-lane WMMA B-fragment layout:
//   Wp[((kt*nTiles + nt)*32 + lane)*16 + e] = W[k][n]
//   lane<16 : n = nt*16+lane,    k = kt*32 + e        (e = 0..15)
//   lane>=16: n = nt*16+lane-16, k = kt*32 + 16 + e
// so a wave loads one 16x16x32 B fragment as two contiguous b128 loads.
// ---------------------------------------------------------------------------
__global__ void k_pack_w(const float* __restrict__ W, bf16* __restrict__ Wp, int NC) {
  int nTiles = NC >> 4;
  int total  = K_ * NC;
  int stride = gridDim.x * blockDim.x;
  for (int o = blockIdx.x * blockDim.x + threadIdx.x; o < total; o += stride) {
    int e    = o & 15;
    int lane = (o >> 4) & 31;
    int t    = o >> 9;
    int nt   = t % nTiles;
    int kt   = t / nTiles;
    int n    = nt * 16 + (lane & 15);
    int k    = kt * 32 + ((lane >> 4) << 4) + e;
    Wp[o] = (bf16)W[k * NC + n];
  }
}

// ---------------------------------------------------------------------------
// WMMA bf16 GEMM:  out[M x NC] = A[M x 256] * W[256 x NC]  (+ bias, fp32 out)
// Block: 256 threads = 8 waves stacked in M.  Wave tile: 16(M) x 64(N).
// The block's 32KB W panel (8 kt-rows x 4 fragments x 1KB) is staged into
// dynamic LDS by the Tensor Data Mover (one tensor_load_to_lds per block,
// TENSORcnt-tracked); the mainloop reads B fragments via ds_load_b128 and
// streams A rows from global as b128 pairs.
// ---------------------------------------------------------------------------
__device__ __forceinline__ v8f wmma_bf16(v16bf a, v16bf b, v8f c) {
  return __builtin_amdgcn_wmma_f32_16x16x32_bf16(false, a, false, b,
                                                 (short)0, c, false, false);
}

template <int NC, bool F32OUT>
__global__ __launch_bounds__(256)
void k_gemm(const bf16* __restrict__ A, const bf16* __restrict__ Wp,
            bf16* __restrict__ outb, float* __restrict__ outf,
            const float* __restrict__ bias) {
  extern __shared__ bf16 sW[];   // 8 * 2048 bf16 = 32 KB, LDS offset 0

  const int lane = threadIdx.x & 31;
  const int wid  = threadIdx.x >> 5;
  const int m0   = (blockIdx.x * 8 + wid) * 16;
  const int n0   = blockIdx.y * 64;
  const int hi   = lane >> 4;
  const int l15  = lane & 15;
  const int nTiles = NC >> 4;

  // ---- TDM: DMA the W panel for this block's n0 into LDS (wave 0 only) ----
  if (wid == 0) {
    unsigned long long gaddr =
        (unsigned long long)(const void*)(Wp + (size_t)(n0 >> 4) * 512);
    // D# group 0: count=1, lds_addr=0, global_addr, type=2
    u32x4 g0;
    g0[0] = 1u;
    g0[1] = 0u;                                    // lds_addr (dynamic LDS base)
    g0[2] = (unsigned)gaddr;
    g0[3] = (unsigned)(gaddr >> 32) | (2u << 30);  // addr[56:32] | type=2
    // D# group 1: 2D tile, data_size=4B.
    // tensor_dim0 = tile_dim0 = 1024 (4B units = 2048 bf16 = 4 fragments)
    // tensor_dim1 = tile_dim1 = 8 (kt rows), dim0 stride = nTiles*256 (4B units)
    const unsigned S0 = (unsigned)nTiles * 256u;
    u32x8 g1;
    g1[0] = (2u << 16);          // data_size = 4 bytes; no multicast/pad/iterate
    g1[1] = (1024u << 16);       // tensor_dim0[15:0] << 16 | atomic_addr(0)
    g1[2] = (8u << 16);          // tensor_dim0[31:16]=0 | tensor_dim1[15:0]=8
    g1[3] = (1024u << 16);       // tensor_dim1[31:16]=0 | tile_dim0=1024
    g1[4] = 8u;                  // tile_dim1=8 | tile_dim2=0
    g1[5] = S0;                  // tensor_dim0_stride[31:0]
    g1[6] = 0u;                  // stride hi / dim1 stride lo
    g1[7] = 0u;                  // dim1 stride hi
    asm volatile("tensor_load_to_lds %0, %1" :: "s"(g0), "s"(g1) : "memory");
    __builtin_amdgcn_s_wait_tensorcnt((unsigned short)0);
  }
  __syncthreads();

  const bf16* aRow = A + (size_t)(m0 + l15) * K_ + hi * 16;

  v8f acc[4] = {v8f{}, v8f{}, v8f{}, v8f{}};

#pragma unroll
  for (int kt = 0; kt < K_ / 32; ++kt) {
    v8bf alo = *(const v8bf*)(aRow + kt * 32);
    v8bf ahi = *(const v8bf*)(aRow + kt * 32 + 8);
    v16bf a = __builtin_shufflevector(alo, ahi, 0, 1, 2, 3, 4, 5, 6, 7,
                                      8, 9, 10, 11, 12, 13, 14, 15);
    const bf16* wBase = sW + ((kt * 4) * 32 + lane) * 16;
#pragma unroll
    for (int f = 0; f < 4; ++f) {
      v8bf blo = *(const v8bf*)(wBase + f * 512);       // ds_load_b128
      v8bf bhi = *(const v8bf*)(wBase + f * 512 + 8);   // ds_load_b128
      v16bf b = __builtin_shufflevector(blo, bhi, 0, 1, 2, 3, 4, 5, 6, 7,
                                        8, 9, 10, 11, 12, 13, 14, 15);
      acc[f] = wmma_bf16(a, b, acc[f]);
    }
  }

  // C/D layout: VGPR r -> row m0 + r + 8*hi ; col = n0 + f*16 + (lane&15)
#pragma unroll
  for (int f = 0; f < 4; ++f) {
    int col = n0 + f * 16 + l15;
#pragma unroll
    for (int r = 0; r < 8; ++r) {
      int row = m0 + hi * 8 + r;
      float val = acc[f][r];
      if constexpr (F32OUT) {
        outf[(size_t)row * NC + col] = val + bias[col];
      } else {
        outb[(size_t)row * NC + col] = (bf16)val;
      }
    }
  }
}

// ---------------------------------------------------------------------------
// Hydra reduce: per (b,h,n) compute rsqrt(|q|^2) and accumulate kn*v over n.
// One wave per pixel-slice: lane = channel (CH=32 == wave32).
// grid.x = B*NH, grid.y = N/256.  Block = 256 threads (8 waves x 32 pixels).
// ---------------------------------------------------------------------------
__global__ __launch_bounds__(256)
void k_reduce(const bf16* __restrict__ qkv, float* __restrict__ rinvq,
              float* __restrict__ kv) {
  const int bh   = blockIdx.x;
  const int b    = bh >> 3;
  const int h    = bh & 7;
  const int lane = threadIdx.x & 31;
  const int wid  = threadIdx.x >> 5;
  const int nbase = blockIdx.y * 256;

  float acc = 0.0f;
#pragma unroll 4
  for (int i = 0; i < 32; ++i) {
    int n = nbase + wid + i * 8;
    size_t roff = (size_t)(b * N_ + n) * 768 + h * 32 + lane;
    float qv = (float)qkv[roff];
    float kk = (float)qkv[roff + 256];
    float vv = (float)qkv[roff + 512];
    float qss = qv * qv;
    float kss = kk * kk;
#pragma unroll
    for (int m = 16; m; m >>= 1) {
      qss += __shfl_xor(qss, m, 32);
      kss += __shfl_xor(kss, m, 32);
    }
    if (lane == 0) rinvq[(size_t)bh * N_ + n] = rsqrtf(qss);
    acc += kk * rsqrtf(kss) * vv;
  }

  __shared__ float red[256];
  red[threadIdx.x] = acc;
  __syncthreads();
  if (wid == 0) {
    float s = 0.0f;
#pragma unroll
    for (int w = 0; w < 8; ++w) s += red[w * 32 + lane];
    atomicAdd(&kv[bh * 32 + lane], s);
  }
}

// ---------------------------------------------------------------------------
// Repack v from qkv rows (B,N,768)[512..767] into planar (B,C,H,W) bf16.
// LDS 64x64 tile transpose for coalesced reads AND writes.
// grid: x = N/64, y = C/64, z = B.
// ---------------------------------------------------------------------------
__global__ __launch_bounds__(256)
void k_repack_v(const bf16* __restrict__ qkv, bf16* __restrict__ vplan) {
  __shared__ bf16 t[64][65];
  const int b  = blockIdx.z;
  const int n0 = blockIdx.x * 64;
  const int c0 = blockIdx.y * 64;
  const int tc = threadIdx.x & 63;
  const int tr = threadIdx.x >> 6;   // 0..3
#pragma unroll
  for (int i = 0; i < 16; ++i) {
    int nn = tr + i * 4;
    t[nn][tc] = qkv[(size_t)(b * N_ + n0 + nn) * 768 + 512 + c0 + tc];
  }
  __syncthreads();
#pragma unroll
  for (int i = 0; i < 16; ++i) {
    int cc = tr + i * 4;
    vplan[(size_t)(b * 256 + c0 + cc) * N_ + n0 + tc] = t[tc][cc];
  }
}

// ---------------------------------------------------------------------------
// Depthwise conv (3x3 / 5x5 / 7x7 per head group) fused with hydra combine:
//   attn[b,n,c] = SCALE * (q * rinvq) * kv  +  q * (dwconv(v) + bias)
// grid: x = W/32, y = H/32, z = B*C.  32x32 output tile with halo in LDS.
// ---------------------------------------------------------------------------
__global__ __launch_bounds__(256)
void k_conv_combine(const bf16* __restrict__ vplan, const bf16* __restrict__ qkv,
                    const float* __restrict__ rinvq, const float* __restrict__ kv,
                    const float* __restrict__ w3, const float* __restrict__ b3,
                    const float* __restrict__ w5, const float* __restrict__ b5,
                    const float* __restrict__ w7, const float* __restrict__ b7,
                    bf16* __restrict__ attn) {
  const int bc = blockIdx.z;
  const int b  = bc >> 8;
  const int c  = bc & 255;
  const int h  = c >> 5;
  const int ch = c & 31;

  int ksz, wc, gch;
  const float *wk, *bk;
  if (c < 64)       { ksz = 3; wk = w3; bk = b3; wc = c;       gch = 64; }
  else if (c < 160) { ksz = 5; wk = w5; bk = b5; wc = c - 64;  gch = 96; }
  else              { ksz = 7; wk = w7; bk = b7; wc = c - 160; gch = 96; }
  const int pad = ksz >> 1;
  const int tw  = 32 + 2 * pad;

  __shared__ float tile[38 * 38];
  __shared__ float wsh[49];
  if ((int)threadIdx.x < ksz * ksz)
    wsh[threadIdx.x] = wk[threadIdx.x * gch + wc];

  const bf16* vp = vplan + (size_t)bc * N_;
  const int y0 = blockIdx.y * 32 - pad;
  const int x0 = blockIdx.x * 32 - pad;
  for (int i = threadIdx.x; i < tw * tw; i += 256) {
    int yy = y0 + i / tw;
    int xx = x0 + i % tw;
    float v = 0.0f;
    if (yy >= 0 && yy < HH_ && xx >= 0 && xx < WW_)
      v = (float)vp[yy * WW_ + xx];
    tile[i] = v;
  }
  __syncthreads();

  const float bias = bk[wc];
  const float kvv  = kv[(b * 8 + h) * 32 + ch];

  for (int i = threadIdx.x; i < 1024; i += 256) {
    int y = i >> 5;
    int x = i & 31;
    float s = bias;
    for (int ky = 0; ky < ksz; ++ky)
      for (int kx = 0; kx < ksz; ++kx)
        s += wsh[ky * ksz + kx] * tile[(y + ky) * tw + (x + kx)];
    int gy = blockIdx.y * 32 + y;
    int gx = blockIdx.x * 32 + x;
    int n  = gy * WW_ + gx;
    size_t row = (size_t)(b * N_ + n);
    float qv  = (float)qkv[row * 768 + h * 32 + ch];
    float riq = rinvq[(size_t)(b * 8 + h) * N_ + n];
    float att = SCALE_ * (qv * riq) * kvv + qv * s;
    attn[row * 256 + c] = (bf16)att;
  }
}

// ---------------------------------------------------------------------------
// Launch
// ---------------------------------------------------------------------------
extern "C" void kernel_launch(void* const* d_in, const int* in_sizes, int n_in,
                              void* d_out, int out_size, void* d_ws, size_t ws_size,
                              hipStream_t stream) {
  (void)in_sizes; (void)n_in; (void)out_size; (void)ws_size;
  const float* x      = (const float*)d_in[0];
  const float* w_qkv  = (const float*)d_in[1];
  const float* w_proj = (const float*)d_in[2];
  const float* b_proj = (const float*)d_in[3];
  const float* w3     = (const float*)d_in[4];
  const float* b3     = (const float*)d_in[5];
  const float* w5     = (const float*)d_in[6];
  const float* b5     = (const float*)d_in[7];
  const float* w7     = (const float*)d_in[8];
  const float* b7     = (const float*)d_in[9];
  float* out = (float*)d_out;

  char* ws = (char*)d_ws;
  auto alloc = [&](size_t bytes) -> char* {
    char* p = ws;
    ws += (bytes + 255) & ~(size_t)255;
    return p;
  };
  bf16*  xb      = (bf16*)alloc((size_t)B_ * N_ * C_ * 2);        // 67 MB
  bf16*  wqkv_p  = (bf16*)alloc((size_t)K_ * 768 * 2);
  bf16*  wproj_p = (bf16*)alloc((size_t)K_ * 256 * 2);
  bf16*  qkv     = (bf16*)alloc((size_t)B_ * N_ * 768 * 2);       // 201 MB
  bf16*  vplan   = (bf16*)alloc((size_t)B_ * C_ * N_ * 2);        // 67 MB
  bf16*  attn    = (bf16*)alloc((size_t)B_ * N_ * C_ * 2);        // 67 MB
  float* rinvq   = (float*)alloc((size_t)B_ * NH_ * N_ * 4);      // 4 MB
  float* kvbuf   = (float*)alloc((size_t)B_ * NH_ * CH_ * 4);

  // 1. convert / pack weights
  k_f32_to_bf16<<<4096, 256, 0, stream>>>(x, xb, B_ * N_ * C_);
  k_pack_w<<<768, 256, 0, stream>>>(w_qkv, wqkv_p, 768);
  k_pack_w<<<256, 256, 0, stream>>>(w_proj, wproj_p, 256);
  k_zero<<<8, 256, 0, stream>>>(kvbuf, B_ * NH_ * CH_);

  // 2. qkv = x @ w_qkv  (WMMA bf16, TDM-staged W panel, 32KB dynamic LDS)
  k_gemm<768, false><<<dim3(1024, 12), 256, 32768, stream>>>(xb, wqkv_p, qkv, nullptr, nullptr);

  // 3. hydra reduction (rinvq, kv)
  k_reduce<<<dim3(64, 64), 256, 0, stream>>>(qkv, rinvq, kvbuf);

  // 4. v -> planar ; depthwise conv + combine
  k_repack_v<<<dim3(N_ / 64, C_ / 64, B_), 256, 0, stream>>>(qkv, vplan);
  k_conv_combine<<<dim3(4, 4, B_ * C_), 256, 0, stream>>>(
      vplan, qkv, rinvq, kvbuf, w3, b3, w5, b5, w7, b7, attn);

  // 5. out = attn @ w_proj + b_proj  (WMMA bf16, fp32 out)
  k_gemm<256, true><<<dim3(1024, 4), 256, 32768, stream>>>(attn, wproj_p, nullptr, out, b_proj);
}
